// DistanceScoreModel_10428180595296
// MI455X (gfx1250) — matrix operand order repaired and checked by reading, not compiled, not run on hardware
//
#include <hip/hip_runtime.h>

#define N_NODES 10000
#define N_EDGES 320000
#define HDIM 256
#define NUM_LAYERS 10

typedef __attribute__((ext_vector_type(16))) __bf16 v16bf;
typedef __attribute__((ext_vector_type(16))) unsigned short v16us;
typedef __attribute__((ext_vector_type(8))) float v8f;

union Frag16 { v16us u; v16bf b; uint4 q[2]; };

__device__ __forceinline__ unsigned short f2bf(float x) {
  union { float f; unsigned u; } v; v.f = x;
  unsigned r = v.u + 0x7FFFu + ((v.u >> 16) & 1u);
  return (unsigned short)(r >> 16);
}
__device__ __forceinline__ float bf2f(unsigned short s) {
  union { unsigned u; float f; } o; o.u = ((unsigned)s) << 16; return o.f;
}

// ---- CDNA5 async global->LDS copy (ASYNCcnt tracked, bypasses VGPRs) ----
// INST_OFFSET is added to BOTH the LDS and global addresses (ISA 08 section 4.4),
// so one address pair covers a 64B row with 4 immediate-offset instructions.
__device__ __forceinline__ void async_row64(unsigned short* lds_dst, const unsigned short* gsrc) {
  unsigned lds_off = (unsigned)(uintptr_t)lds_dst;  // low 32 bits of generic ptr = LDS offset
  asm volatile(
      "global_load_async_to_lds_b128 %0, %1, off\n\t"
      "global_load_async_to_lds_b128 %0, %1, off offset:16\n\t"
      "global_load_async_to_lds_b128 %0, %1, off offset:32\n\t"
      "global_load_async_to_lds_b128 %0, %1, off offset:48"
      :: "v"(lds_off), "v"(gsrc) : "memory");
}
template <int N>
__device__ __forceinline__ void wait_async() {
  asm volatile("s_wait_asynccnt %0" :: "i"(N) : "memory");
}
// Stage a 32-K slab of an N x K (pre-transposed) bf16 weight matrix into LDS
// laid out [n][32]. Each wave issues (NROWS/128)*4 async b128 instructions.
template <int NROWS>
__device__ __forceinline__ void issue_slab(const unsigned short* __restrict__ wT, int K, int k0,
                                           unsigned short* sWT, int tid) {
#pragma unroll
  for (int j = 0; j < NROWS / 128; ++j) {
    int n = j * 128 + tid;
    async_row64(sWT + n * 32, wT + (size_t)n * K + k0);
  }
}

// A fragment: 16x32 bf16 tile, row-major stride lda. Two ds_load_b128 per lane.
// lanes 0-15: K = 0..7 then 16..23 ; lanes 16-31: K = 8..15 then 24..31
__device__ __forceinline__ v16bf make_A(const unsigned short* A, int lda, int lane) {
  int row = lane & 15;
  int kb = (lane & 16) ? 8 : 0;
  Frag16 f;
  f.q[0] = *reinterpret_cast<const uint4*>(A + row * lda + kb);
  f.q[1] = *reinterpret_cast<const uint4*>(A + row * lda + 16 + kb);
  return f.b;
}
// A fragment where every row is the same 32-long vector (broadcast row)
__device__ __forceinline__ v16bf make_A_bcast(const unsigned short* Avec, int lane) {
  int kb = (lane & 16) ? 8 : 0;
  Frag16 f;
  f.q[0] = *reinterpret_cast<const uint4*>(Avec + kb);
  f.q[1] = *reinterpret_cast<const uint4*>(Avec + 16 + kb);
  return f.b;
}
// B fragment from K-major (transposed) LDS tile: Bt[col][32] for one 16-col tile.
// lanes 0-15: K=0..7 then 16..23 ; lanes 16-31: K=8..15 then 24..31 ; col = lane & 15. Two ds_load_b128.
__device__ __forceinline__ v16bf make_Bt(const unsigned short* Bt, int lane) {
  int col = lane & 15;
  int kb = (lane & 16) ? 8 : 0;
  Frag16 f;
  f.q[0] = *reinterpret_cast<const uint4*>(Bt + col * 32 + kb);
  f.q[1] = *reinterpret_cast<const uint4*>(Bt + col * 32 + 16 + kb);
  return f.b;
}

#define WMMA_BF16(A, B, C) \
  __builtin_amdgcn_wmma_f32_16x16x32_bf16(false, (A), false, (B), (short)0, (C), false, false)

// ---------------- small utility kernels ----------------

// Convert fp32 K x N weight (batched) to bf16 transposed N x K layout.
__global__ void cvt_t_kernel(const float* __restrict__ src, unsigned short* __restrict__ dst,
                             int K, int N, int total) {
  int i = blockIdx.x * blockDim.x + threadIdx.x;
  if (i >= total) return;
  int kn = K * N;
  int l = i / kn, r = i % kn;
  int k = r / N, n = r % N;
  dst[(size_t)l * kn + (size_t)n * K + k] = f2bf(src[i]);
}

__global__ void zero_kernel(float* __restrict__ p, int n) {
  int i = blockIdx.x * blockDim.x + threadIdx.x;
  if (i < n) p[i] = 0.f;
}

__global__ void init_h_kernel(const float* __restrict__ node_emb,
                              const int* __restrict__ atom_type,
                              float* __restrict__ h) {
  int n = blockIdx.x, c = threadIdx.x;
  h[(size_t)n * HDIM + c] = node_emb[(size_t)atom_type[n] * HDIM + c];
}

__global__ __launch_bounds__(HDIM) void sigma_kernel(
    const float* __restrict__ sigma,
    const float* __restrict__ w1, const float* __restrict__ b1,
    const float* __restrict__ w2, const float* __restrict__ b2,
    unsigned short* __restrict__ sig_bf) {
  __shared__ float sH[HDIM];
  int j = threadIdx.x;
  float lsg = logf(fmaxf(sigma[0], 1e-12f));
  sH[j] = fmaxf(lsg * w1[j] + b1[j], 0.f);
  __syncthreads();
  float acc = b2[j];
  for (int k = 0; k < HDIM; ++k) acc += sH[k] * w2[k * HDIM + j];
  sig_bf[j] = f2bf(acc);
}

// ---------------- edge distance MLP + edge_feat ----------------
__global__ __launch_bounds__(128) void edge_dist_kernel(
    const float* __restrict__ pos,
    const float* __restrict__ dist_w1, const float* __restrict__ dist_b1,
    const unsigned short* __restrict__ w2T, const float* __restrict__ dist_b2,
    const float* __restrict__ edge_emb,
    const int* __restrict__ edge_type, const int* __restrict__ edge_index,
    const unsigned char* __restrict__ edge_mask,
    unsigned short* __restrict__ dist_emb_bf, unsigned short* __restrict__ edge_feat_bf) {
  __shared__ __align__(16) unsigned short sH1[16 * HDIM];
  __shared__ __align__(16) unsigned short sW0[HDIM * 32];
  __shared__ __align__(16) unsigned short sW1[HDIM * 32];
  __shared__ float sDist[16];
  __shared__ float sMask[16];
  __shared__ int sEtype[16];
  int tid = threadIdx.x, lane = tid & 31, wv = tid >> 5;
  int e0 = blockIdx.x * 16;
  __builtin_amdgcn_s_wait_tensorcnt(0);
  if (tid < 16) {
    int e = e0 + tid;
    int s = edge_index[e], d = edge_index[N_EDGES + e];
    float dx = pos[3 * s] - pos[3 * d];
    float dy = pos[3 * s + 1] - pos[3 * d + 1];
    float dz = pos[3 * s + 2] - pos[3 * d + 2];
    float m = edge_mask[e] ? 1.f : 0.f;
    sMask[tid] = m;
    sEtype[tid] = edge_type[e];
    sDist[tid] = sqrtf(dx * dx + dy * dy + dz * dz) * m;
  }
  issue_slab<HDIM>(w2T, HDIM, 0, sW0, tid);
  __syncthreads();
  for (int i = tid; i < 16 * HDIM; i += 128) {
    int r = i >> 8, j = i & 255;
    sH1[i] = f2bf(fmaxf(sDist[r] * dist_w1[j] + dist_b1[j], 0.f));
  }
  v8f c0 = {}, c1 = {}, c2 = {}, c3 = {};
  int ncol0 = wv * 64;
  for (int k0 = 0; k0 < HDIM; k0 += 32) {
    unsigned short* cur = ((k0 >> 5) & 1) ? sW1 : sW0;
    unsigned short* nxt = ((k0 >> 5) & 1) ? sW0 : sW1;
    if (k0 + 32 < HDIM) { issue_slab<HDIM>(w2T, HDIM, k0 + 32, nxt, tid); wait_async<8>(); }
    else                { wait_async<0>(); }
    __syncthreads();
    v16bf a = make_A(sH1 + k0, HDIM, lane);
    c0 = WMMA_BF16(a, make_Bt(cur + (ncol0     ) * 32, lane), c0);
    c1 = WMMA_BF16(a, make_Bt(cur + (ncol0 + 16) * 32, lane), c1);
    c2 = WMMA_BF16(a, make_Bt(cur + (ncol0 + 32) * 32, lane), c2);
    c3 = WMMA_BF16(a, make_Bt(cur + (ncol0 + 48) * 32, lane), c3);
    __syncthreads();
  }
  v8f acc[4] = {c0, c1, c2, c3};
  int col = lane & 15;
  int r0 = (lane & 16) ? 8 : 0;
#pragma unroll
  for (int t = 0; t < 4; ++t) {
    int colg = ncol0 + t * 16 + col;
    float bias = dist_b2[colg];
#pragma unroll
    for (int i = 0; i < 8; ++i) {
      int r = r0 + i;
      size_t e = (size_t)(e0 + r);
      float val = (acc[t][i] + bias) * sMask[r];
      dist_emb_bf[e * HDIM + colg] = f2bf(val);
      float ef = edge_emb[(size_t)sEtype[r] * HDIM + colg] * val;
      edge_feat_bf[e * HDIM + colg] = f2bf(ef);
    }
  }
}

// ---------------- message + segment-sum ----------------
__global__ __launch_bounds__(HDIM) void msg_agg_kernel(
    const float* __restrict__ h, const unsigned short* __restrict__ edge_feat_bf,
    const int* __restrict__ edge_index, const unsigned char* __restrict__ edge_mask,
    float* __restrict__ agg) {
  int e = blockIdx.x;
  if (!edge_mask[e]) return;
  int c = threadIdx.x;
  int s = edge_index[e], d = edge_index[N_EDGES + e];
  float msg = h[(size_t)s * HDIM + c] + bf2f(edge_feat_bf[(size_t)e * HDIM + c]);
  msg = fmaxf(msg, 0.f);
  atomicAdd(&agg[(size_t)d * HDIM + c], msg);
}

// ---------------- GIN node MLP (two fused WMMA GEMMs) ----------------
__global__ __launch_bounds__(128) void node_mlp_kernel(
    const float* __restrict__ h, const float* __restrict__ agg,
    const unsigned short* __restrict__ w1T, const float* __restrict__ b1,
    const unsigned short* __restrict__ w2T, const float* __restrict__ b2,
    float* __restrict__ h_out, int apply_relu) {
  __shared__ __align__(16) unsigned short sX[16 * HDIM];
  __shared__ __align__(16) unsigned short sMid[16 * HDIM];
  __shared__ __align__(16) unsigned short sW0[HDIM * 32];
  __shared__ __align__(16) unsigned short sW1[HDIM * 32];
  int tid = threadIdx.x, lane = tid & 31, wv = tid >> 5;
  int n0 = blockIdx.x * 16;
  issue_slab<HDIM>(w1T, HDIM, 0, sW0, tid);
  for (int i = tid; i < 16 * HDIM; i += 128) {
    int r = i >> 8, j = i & 255;
    size_t idx = (size_t)(n0 + r) * HDIM + j;
    sX[i] = f2bf(h[idx] + agg[idx]);
  }
  int ncol0 = wv * 64;
  int col = lane & 15;
  int r0 = (lane & 16) ? 8 : 0;
  v8f zero = {};
  v8f c0 = zero, c1 = zero, c2 = zero, c3 = zero;
  for (int k0 = 0; k0 < HDIM; k0 += 32) {
    unsigned short* cur = ((k0 >> 5) & 1) ? sW1 : sW0;
    unsigned short* nxt = ((k0 >> 5) & 1) ? sW0 : sW1;
    if (k0 + 32 < HDIM) { issue_slab<HDIM>(w1T, HDIM, k0 + 32, nxt, tid); wait_async<8>(); }
    else                { wait_async<0>(); }
    __syncthreads();
    v16bf a = make_A(sX + k0, HDIM, lane);
    c0 = WMMA_BF16(a, make_Bt(cur + (ncol0     ) * 32, lane), c0);
    c1 = WMMA_BF16(a, make_Bt(cur + (ncol0 + 16) * 32, lane), c1);
    c2 = WMMA_BF16(a, make_Bt(cur + (ncol0 + 32) * 32, lane), c2);
    c3 = WMMA_BF16(a, make_Bt(cur + (ncol0 + 48) * 32, lane), c3);
    __syncthreads();
  }
  {
    v8f acc[4] = {c0, c1, c2, c3};
#pragma unroll
    for (int t = 0; t < 4; ++t) {
      int colg = ncol0 + t * 16 + col;
      float bias = b1[colg];
#pragma unroll
      for (int i = 0; i < 8; ++i)
        sMid[(r0 + i) * HDIM + colg] = f2bf(fmaxf(acc[t][i] + bias, 0.f));
    }
  }
  c0 = zero; c1 = zero; c2 = zero; c3 = zero;
  issue_slab<HDIM>(w2T, HDIM, 0, sW0, tid);
  for (int k0 = 0; k0 < HDIM; k0 += 32) {
    unsigned short* cur = ((k0 >> 5) & 1) ? sW1 : sW0;
    unsigned short* nxt = ((k0 >> 5) & 1) ? sW0 : sW1;
    if (k0 + 32 < HDIM) { issue_slab<HDIM>(w2T, HDIM, k0 + 32, nxt, tid); wait_async<8>(); }
    else                { wait_async<0>(); }
    __syncthreads();
    v16bf a = make_A(sMid + k0, HDIM, lane);
    c0 = WMMA_BF16(a, make_Bt(cur + (ncol0     ) * 32, lane), c0);
    c1 = WMMA_BF16(a, make_Bt(cur + (ncol0 + 16) * 32, lane), c1);
    c2 = WMMA_BF16(a, make_Bt(cur + (ncol0 + 32) * 32, lane), c2);
    c3 = WMMA_BF16(a, make_Bt(cur + (ncol0 + 48) * 32, lane), c3);
    __syncthreads();
  }
  {
    v8f acc[4] = {c0, c1, c2, c3};
#pragma unroll
    for (int t = 0; t < 4; ++t) {
      int colg = ncol0 + t * 16 + col;
      float bias = b2[colg];
#pragma unroll
      for (int i = 0; i < 8; ++i) {
        float v = acc[t][i] + bias;
        if (apply_relu) v = fmaxf(v, 0.f);
        h_out[(size_t)(n0 + r0 + i) * HDIM + colg] = v;
      }
    }
  }
}

// ---------------- fused output MLP3 over the 768-wide concat ----------------
__global__ __launch_bounds__(128) void out_mlp_kernel(
    const float* __restrict__ h,
    const unsigned short* __restrict__ dist_emb_bf,
    const unsigned short* __restrict__ sig_bf,
    const unsigned short* __restrict__ w1T, const float* __restrict__ b1,
    const unsigned short* __restrict__ w2T, const float* __restrict__ b2,
    const float* __restrict__ w3, const float* __restrict__ b3,
    const int* __restrict__ edge_index, const unsigned char* __restrict__ edge_mask,
    const float* __restrict__ sigma,
    float* __restrict__ out) {
  __shared__ __align__(16) unsigned short sHC[16 * HDIM];
  __shared__ __align__(16) unsigned short sDE[16 * HDIM];
  __shared__ __align__(16) unsigned short sSG[HDIM];
  __shared__ __align__(16) unsigned short sMid[16 * HDIM];
  __shared__ __align__(16) unsigned short sMid2[16 * 128];
  __shared__ __align__(16) unsigned short sW0[HDIM * 32];
  __shared__ __align__(16) unsigned short sW1[HDIM * 32];
  __shared__ float sMask[16];
  __shared__ int sSrc[16], sDst[16];
  int tid = threadIdx.x, lane = tid & 31, wv = tid >> 5;
  int e0 = blockIdx.x * 16;
  if (tid < 16) {
    int e = e0 + tid;
    sSrc[tid] = edge_index[e];
    sDst[tid] = edge_index[N_EDGES + e];
    sMask[tid] = edge_mask[e] ? 1.f : 0.f;
  }
  issue_slab<HDIM>(w1T, 3 * HDIM, 0, sW0, tid);
  for (int i = tid; i < HDIM; i += 128) sSG[i] = sig_bf[i];
  __syncthreads();
  for (int i = tid; i < 16 * HDIM; i += 128) {
    int r = i >> 8, j = i & 255;
    float hc = h[(size_t)sSrc[r] * HDIM + j] * h[(size_t)sDst[r] * HDIM + j] * sMask[r];
    sHC[i] = f2bf(hc);
    sDE[i] = dist_emb_bf[(size_t)(e0 + r) * HDIM + j];
  }
  int ncol0 = wv * 64;
  int col = lane & 15;
  int r0 = (lane & 16) ? 8 : 0;
  v8f zero = {};
  v8f c0 = zero, c1 = zero, c2 = zero, c3 = zero;
  // GEMM1: (16 x 768) x (768 x 256); A assembled from 3 sources, never concat'd
  for (int k0 = 0; k0 < 3 * HDIM; k0 += 32) {
    unsigned short* cur = ((k0 >> 5) & 1) ? sW1 : sW0;
    unsigned short* nxt = ((k0 >> 5) & 1) ? sW0 : sW1;
    if (k0 + 32 < 3 * HDIM) { issue_slab<HDIM>(w1T, 3 * HDIM, k0 + 32, nxt, tid); wait_async<8>(); }
    else                    { wait_async<0>(); }
    __syncthreads();
    v16bf a;
    if (k0 < HDIM)          a = make_A(sHC + k0, HDIM, lane);
    else if (k0 < 2 * HDIM) a = make_A(sDE + (k0 - HDIM), HDIM, lane);
    else                    a = make_A_bcast(sSG + (k0 - 2 * HDIM), lane);
    c0 = WMMA_BF16(a, make_Bt(cur + (ncol0     ) * 32, lane), c0);
    c1 = WMMA_BF16(a, make_Bt(cur + (ncol0 + 16) * 32, lane), c1);
    c2 = WMMA_BF16(a, make_Bt(cur + (ncol0 + 32) * 32, lane), c2);
    c3 = WMMA_BF16(a, make_Bt(cur + (ncol0 + 48) * 32, lane), c3);
    __syncthreads();
  }
  {
    v8f acc[4] = {c0, c1, c2, c3};
#pragma unroll
    for (int t = 0; t < 4; ++t) {
      int colg = ncol0 + t * 16 + col;
      float bias = b1[colg];
#pragma unroll
      for (int i = 0; i < 8; ++i)
        sMid[(r0 + i) * HDIM + colg] = f2bf(fmaxf(acc[t][i] + bias, 0.f));
    }
  }
  // GEMM2: (16 x 256) x (256 x 128); 8 N-tiles -> 2 per wave
  v8f d0 = zero, d1 = zero;
  int ncol2 = wv * 32;
  issue_slab<128>(w2T, HDIM, 0, sW0, tid);
  for (int k0 = 0; k0 < HDIM; k0 += 32) {
    unsigned short* cur = ((k0 >> 5) & 1) ? sW1 : sW0;
    unsigned short* nxt = ((k0 >> 5) & 1) ? sW0 : sW1;
    if (k0 + 32 < HDIM) { issue_slab<128>(w2T, HDIM, k0 + 32, nxt, tid); wait_async<4>(); }
    else                { wait_async<0>(); }
    __syncthreads();
    v16bf a = make_A(sMid + k0, HDIM, lane);
    d0 = WMMA_BF16(a, make_Bt(cur + (ncol2     ) * 32, lane), d0);
    d1 = WMMA_BF16(a, make_Bt(cur + (ncol2 + 16) * 32, lane), d1);
    __syncthreads();
  }
  {
    v8f acc[2] = {d0, d1};
#pragma unroll
    for (int t = 0; t < 2; ++t) {
      int colg = ncol2 + t * 16 + col;
      float bias = b2[colg];
#pragma unroll
      for (int i = 0; i < 8; ++i)
        sMid2[(r0 + i) * 128 + colg] = f2bf(fmaxf(acc[t][i] + bias, 0.f));
    }
  }
  __syncthreads();
  // GEMM3: (16 x 128) x (128 x 1) as per-edge dot
  if (tid < 16) {
    float accv = b3[0];
    for (int k = 0; k < 128; ++k) accv += bf2f(sMid2[tid * 128 + k]) * w3[k];
    out[e0 + tid] = accv * sMask[tid] / sigma[0];
  }
}

extern "C" void kernel_launch(void* const* d_in, const int* in_sizes, int n_in,
                              void* d_out, int out_size, void* d_ws, size_t ws_size,
                              hipStream_t stream) {
  const float* pos      = (const float*)d_in[0];
  const float* sigma    = (const float*)d_in[1];
  const float* node_emb = (const float*)d_in[2];
  const float* edge_emb = (const float*)d_in[3];
  const float* dist_w1  = (const float*)d_in[4];
  const float* dist_b1  = (const float*)d_in[5];
  const float* dist_w2  = (const float*)d_in[6];
  const float* dist_b2  = (const float*)d_in[7];
  const float* sig_w1   = (const float*)d_in[8];
  const float* sig_b1   = (const float*)d_in[9];
  const float* sig_w2   = (const float*)d_in[10];
  const float* sig_b2   = (const float*)d_in[11];
  const float* gin_w1   = (const float*)d_in[12];
  const float* gin_b1   = (const float*)d_in[13];
  const float* gin_w2   = (const float*)d_in[14];
  const float* gin_b2   = (const float*)d_in[15];
  const float* out_w1   = (const float*)d_in[16];
  const float* out_b1   = (const float*)d_in[17];
  const float* out_w2   = (const float*)d_in[18];
  const float* out_b2   = (const float*)d_in[19];
  const float* out_w3   = (const float*)d_in[20];
  const float* out_b3   = (const float*)d_in[21];
  const int* atom_type  = (const int*)d_in[22];
  const int* edge_type  = (const int*)d_in[23];
  const int* edge_index = (const int*)d_in[24];
  const unsigned char* edge_mask = (const unsigned char*)d_in[25];
  (void)in_sizes; (void)n_in; (void)out_size; (void)ws_size;

  char* ws = (char*)d_ws;
  size_t off = 0;
  auto take = [&](size_t bytes) -> char* {
    char* p = ws + off;
    off = (off + bytes + 255) & ~(size_t)255;
    return p;
  };
  unsigned short* w2T    = (unsigned short*)take((size_t)HDIM * HDIM * 2);
  unsigned short* ginw1T = (unsigned short*)take((size_t)NUM_LAYERS * HDIM * HDIM * 2);
  unsigned short* ginw2T = (unsigned short*)take((size_t)NUM_LAYERS * HDIM * HDIM * 2);
  unsigned short* outw1T = (unsigned short*)take((size_t)3 * HDIM * HDIM * 2);
  unsigned short* outw2T = (unsigned short*)take((size_t)HDIM * 128 * 2);
  unsigned short* sigbf  = (unsigned short*)take((size_t)HDIM * 2);
  float* h_a = (float*)take((size_t)N_NODES * HDIM * 4);
  float* h_b = (float*)take((size_t)N_NODES * HDIM * 4);
  float* agg = (float*)take((size_t)N_NODES * HDIM * 4);
  unsigned short* distEmbBf  = (unsigned short*)take((size_t)N_EDGES * HDIM * 2);
  unsigned short* edgeFeatBf = (unsigned short*)take((size_t)N_EDGES * HDIM * 2);

  cvt_t_kernel<<<(HDIM * HDIM + 255) / 256, 256, 0, stream>>>(dist_w2, w2T, HDIM, HDIM, HDIM * HDIM);
  cvt_t_kernel<<<(NUM_LAYERS * HDIM * HDIM + 255) / 256, 256, 0, stream>>>(gin_w1, ginw1T, HDIM, HDIM, NUM_LAYERS * HDIM * HDIM);
  cvt_t_kernel<<<(NUM_LAYERS * HDIM * HDIM + 255) / 256, 256, 0, stream>>>(gin_w2, ginw2T, HDIM, HDIM, NUM_LAYERS * HDIM * HDIM);
  cvt_t_kernel<<<(3 * HDIM * HDIM + 255) / 256, 256, 0, stream>>>(out_w1, outw1T, 3 * HDIM, HDIM, 3 * HDIM * HDIM);
  cvt_t_kernel<<<(HDIM * 128 + 255) / 256, 256, 0, stream>>>(out_w2, outw2T, HDIM, 128, HDIM * 128);

  sigma_kernel<<<1, HDIM, 0, stream>>>(sigma, sig_w1, sig_b1, sig_w2, sig_b2, sigbf);
  init_h_kernel<<<N_NODES, HDIM, 0, stream>>>(node_emb, atom_type, h_a);
  edge_dist_kernel<<<N_EDGES / 16, 128, 0, stream>>>(
      pos, dist_w1, dist_b1, w2T, dist_b2, edge_emb, edge_type, edge_index,
      edge_mask, distEmbBf, edgeFeatBf);

  float* hc = h_a;
  float* hn = h_b;
  for (int l = 0; l < NUM_LAYERS; ++l) {
    zero_kernel<<<(N_NODES * HDIM + 255) / 256, 256, 0, stream>>>(agg, N_NODES * HDIM);
    msg_agg_kernel<<<N_EDGES, HDIM, 0, stream>>>(hc, edgeFeatBf, edge_index, edge_mask, agg);
    node_mlp_kernel<<<N_NODES / 16, 128, 0, stream>>>(
        hc, agg,
        ginw1T + (size_t)l * HDIM * HDIM, gin_b1 + (size_t)l * HDIM,
        ginw2T + (size_t)l * HDIM * HDIM, gin_b2 + (size_t)l * HDIM,
        hn, (l < NUM_LAYERS - 1) ? 1 : 0);
    float* t = hc; hc = hn; hn = t;
  }

  out_mlp_kernel<<<N_EDGES / 16, 128, 0, stream>>>(
      hc, distEmbBf, sigbf, outw1T, out_b1, outw2T, out_b2, out_w3, out_b3,
      edge_index, edge_mask, sigma, (float*)d_out);
}
